// LSTM_Multi_Head_Attention_77893526880425
// MI455X (gfx1250) — matrix operand order repaired and checked by reading, not compiled, not run on hardware
//
#include <hip/hip_runtime.h>

// ---------------- problem constants ----------------
#define Bn   32
#define Sn   512
#define Hn   512
#define H4   2048
#define Dn   300
#define K0P  320        // 300 padded to multiple of 32
#define BSn  (Bn*Sn)    // 16384
#define NHn  8
#define HDn  128
#define DM   1024       // 2H

typedef __bf16 bf16;
typedef __attribute__((ext_vector_type(16))) __bf16 v16bf;
typedef __attribute__((ext_vector_type(8)))  float  v8f;
typedef __attribute__((ext_vector_type(4)))  unsigned int u32x4;

union ABv { v16bf v; u32x4 q[2]; };

__device__ __forceinline__ bf16 f2bf(float f) {
  unsigned u = __builtin_bit_cast(unsigned, f);
  unsigned r = (u + 0x7fffu + ((u >> 16) & 1u)) >> 16;
  unsigned short h = (unsigned short)r;
  return __builtin_bit_cast(bf16, h);
}

// ---------------- weight prep: bf16 + gate interleave (col = 4*j + gate) ----------------
__global__ void prep_lstm(const float* __restrict__ Wih, const float* __restrict__ Whh,
                          const float* __restrict__ bih, const float* __restrict__ bhh,
                          bf16* __restrict__ WihO, bf16* __restrict__ WhhO,
                          float* __restrict__ biasO, int Kin, int Kpad) {
  size_t tot1 = (size_t)H4 * Kpad;
  size_t tot2 = (size_t)H4 * Hn;
  size_t tot  = tot1 + tot2 + H4;
  size_t stride = (size_t)gridDim.x * blockDim.x;
  for (size_t idx = (size_t)blockIdx.x * blockDim.x + threadIdx.x; idx < tot; idx += stride) {
    if (idx < tot1) {
      int cc = (int)(idx / Kpad), k = (int)(idx % Kpad);
      int g = cc & 3, j = cc >> 2;
      float v = (k < Kin) ? Wih[(size_t)(g * Hn + j) * Kin + k] : 0.f;
      WihO[idx] = f2bf(v);
    } else if (idx < tot1 + tot2) {
      size_t r = idx - tot1;
      int cc = (int)(r / Hn), k = (int)(r % Hn);
      int g = cc & 3, j = cc >> 2;
      WhhO[r] = f2bf(Whh[(size_t)(g * Hn + j) * Hn + k]);
    } else {
      int cc = (int)(idx - tot1 - tot2);
      int g = cc & 3, j = cc >> 2;
      biasO[cc] = bih[g * Hn + j] + bhh[g * Hn + j];
    }
  }
}

// attention weight: WO[(h*128+e)*1024 + d] = W[h][d][e]  (transpose d<->e per head)
__global__ void prep_attn(const float* __restrict__ W, bf16* __restrict__ WO) {
  size_t tot = (size_t)DM * DM;
  size_t stride = (size_t)gridDim.x * blockDim.x;
  for (size_t idx = (size_t)blockIdx.x * blockDim.x + threadIdx.x; idx < tot; idx += stride) {
    int cc = (int)(idx >> 10), dd = (int)(idx & 1023);
    int hh = cc >> 7, ee = cc & 127;
    WO[idx] = f2bf(W[((size_t)hh * DM + dd) * HDn + ee]);
  }
}

// ---------------- embedding gather -> bf16, K padded to 320 ----------------
__global__ void embed_gather(const int* __restrict__ x, const float* __restrict__ emb,
                             bf16* __restrict__ X0) {
  size_t tot = (size_t)BSn * K0P;
  size_t stride = (size_t)gridDim.x * blockDim.x;
  for (size_t idx = (size_t)blockIdx.x * blockDim.x + threadIdx.x; idx < tot; idx += stride) {
    int m = (int)(idx / K0P), k = (int)(idx % K0P);   // m = b*S + s
    int tok = x[m];
    float v = (k < Dn) ? emb[(size_t)tok * Dn + k] : 0.f;
    X0[idx] = f2bf(v);
  }
}

__global__ void zero_f32(float* __restrict__ p, int n) {
  int i = blockIdx.x * blockDim.x + threadIdx.x;
  if (i < n) p[i] = 0.f;
}

// ---------------- generic WMMA GEMM: C[m][n] = sum_k A[m][k]*W[n][k] + bias[n] ----------------
// block = 256 (8 waves), WG covers 32x64 of C; each wave one 16x16 tile.
// mapMode: 0 row-major MxN; 1 xg scatter [s][b][N]; 2 QK scatter [bh][s][e]; 3 V scatter [bh][e][s]
__global__ void gemm_wmma(const bf16* __restrict__ A, int lda,
                          const bf16* __restrict__ W, int ldw,
                          const float* __restrict__ bias,
                          float* __restrict__ outF, bf16* __restrict__ outB,
                          int N, int K, int mapMode) {
  int nWgCols = N >> 6;
  int wgN = blockIdx.x % nWgCols;
  int wgM = blockIdx.x / nWgCols;
  int wave = threadIdx.x >> 5, lane = threadIdx.x & 31;
  int mt = wgM * 2 + (wave >> 2);
  int nt = wgN * 4 + (wave & 3);
  int laneR = lane & 15;
  int hi = (lane >= 16);

  const bf16* Ap = A + (size_t)(mt * 16 + laneR) * lda + (hi ? 8 : 0);
  const bf16* Wp = W + (size_t)(nt * 16 + laneR) * ldw + (hi ? 8 : 0);
  v8f acc = {0.f,0.f,0.f,0.f,0.f,0.f,0.f,0.f};
  for (int k0 = 0; k0 < K; k0 += 32) {
    ABv a, b;
    a.q[0] = *(const u32x4*)(Ap + k0);
    a.q[1] = *(const u32x4*)(Ap + k0 + 16);
    b.q[0] = *(const u32x4*)(Wp + k0);
    b.q[1] = *(const u32x4*)(Wp + k0 + 16);
    acc = __builtin_amdgcn_wmma_f32_16x16x32_bf16(false, a.v, false, b.v, (short)0, acc, false, false);
  }
  int n = nt * 16 + laneR;
  float bsv = bias ? bias[n] : 0.f;
#pragma unroll
  for (int i = 0; i < 8; i++) {
    int r = mt * 16 + i + (hi ? 8 : 0);      // r = b*S + s
    float v = acc[i] + bsv;
    size_t idx;
    if (mapMode == 0) {
      idx = (size_t)r * N + n;
    } else if (mapMode == 1) {
      int bb = r >> 9, ss = r & 511;
      idx = ((size_t)ss * Bn + bb) * N + n;
    } else if (mapMode == 2) {
      int bb = r >> 9, ss = r & 511;
      int hh = n >> 7, ee = n & 127;
      idx = (((size_t)bb * NHn + hh) * Sn + ss) * HDn + ee;
    } else {
      int bb = r >> 9, ss = r & 511;
      int hh = n >> 7, ee = n & 127;
      idx = (((size_t)bb * NHn + hh) * HDn + ee) * Sn + ss;
    }
    if (outF) outF[idx] = v; else outB[idx] = f2bf(v);
  }
}

// ---------------- one LSTM time step, both directions (fwd: t, bwd: S-1-t) ----------------
// grid = 64 blocks x 128 thr; dir = blk>>5; each wave: 32x16 tile of g = h_prev @ Whh'^T
// columns gate-interleaved so each wave owns complete i/f/g/o quadruples of 4 units.
__global__ void lstm_step(int t,
                          const float* __restrict__ xgF, const float* __restrict__ xgB,
                          const bf16* __restrict__ WhhF, const bf16* __restrict__ WhhB,
                          bf16* __restrict__ hbuf,    // [2 bufs][2 dirs][32][512] bf16
                          float* __restrict__ cbuf,   // [2 dirs][32][512] f32
                          bf16* __restrict__ lout) {  // [B*S][1024] bf16 (fwd cols 0..511, bwd 512..1023)
  const int dir  = blockIdx.x >> 5;
  const int wgn  = blockIdx.x & 31;
  const int wave = threadIdx.x >> 5, lane = threadIdx.x & 31;
  const int n0   = (wgn * 4 + wave) * 16;
  const int laneR = lane & 15;
  const int hi    = (lane >= 16);

  const bf16*  Whh = dir ? WhhB : WhhF;
  const float* xg  = dir ? xgB : xgF;
  const int    ta  = dir ? (Sn - 1 - t) : t;

  const bf16* hr = hbuf + (size_t)(t & 1) * 32768 + (size_t)dir * 16384;
  bf16*       hw = hbuf + (size_t)((t + 1) & 1) * 32768 + (size_t)dir * 16384;

  const bf16* A0 = hr + (size_t)laneR * Hn + (hi ? 8 : 0);
  const bf16* A1 = A0 + (size_t)16 * Hn;
  const bf16* Wp = Whh + (size_t)(n0 + laneR) * Hn + (hi ? 8 : 0);

  v8f acc0 = {0.f,0.f,0.f,0.f,0.f,0.f,0.f,0.f};
  v8f acc1 = acc0;
  for (int k0 = 0; k0 < Hn; k0 += 32) {
    ABv bm, a0, a1;
    bm.q[0] = *(const u32x4*)(Wp + k0);
    bm.q[1] = *(const u32x4*)(Wp + k0 + 16);
    a0.q[0] = *(const u32x4*)(A0 + k0);
    a0.q[1] = *(const u32x4*)(A0 + k0 + 16);
    a1.q[0] = *(const u32x4*)(A1 + k0);
    a1.q[1] = *(const u32x4*)(A1 + k0 + 16);
    acc0 = __builtin_amdgcn_wmma_f32_16x16x32_bf16(false, a0.v, false, bm.v, (short)0, acc0, false, false);
    acc1 = __builtin_amdgcn_wmma_f32_16x16x32_bf16(false, a1.v, false, bm.v, (short)0, acc1, false, false);
  }

  __shared__ float G[4][32][16];
  const float* xgt = xg + (size_t)ta * Bn * H4;
#pragma unroll
  for (int i = 0; i < 8; i++) {
    int r0 = i + (hi ? 8 : 0);
    G[wave][r0][laneR]      = acc0[i] + xgt[(size_t)r0 * H4 + n0 + laneR];
    G[wave][r0 + 16][laneR] = acc1[i] + xgt[(size_t)(r0 + 16) * H4 + n0 + laneR];
  }
  __syncthreads();

  float* c = cbuf + (size_t)dir * 16384;
  const int j0 = n0 >> 2;
#pragma unroll
  for (int pp = 0; pp < 4; pp++) {
    int bb = lane;               // row (batch)
    int u  = pp;                 // unit within tile
    float gi = G[wave][bb][u * 4 + 0];
    float gf = G[wave][bb][u * 4 + 1];
    float gg = G[wave][bb][u * 4 + 2];
    float go = G[wave][bb][u * 4 + 3];
    int j = j0 + u;
    float cold = c[bb * Hn + j];
    float si = 1.f / (1.f + __expf(-gi));
    float sf = 1.f / (1.f + __expf(-gf));
    float so = 1.f / (1.f + __expf(-go));
    float cn = sf * cold + si * tanhf(gg);
    float hv = so * tanhf(cn);
    c[bb * Hn + j] = cn;
    hw[bb * Hn + j] = f2bf(hv);
    lout[((size_t)bb * Sn + ta) * DM + dir * Hn + j] = f2bf(hv);
  }
}

// ---------------- flash attention: one wave per (b, head, 16-row Q tile) ----------------
// Q,K: [bh][s][e] bf16; Vt: [bh][e][s] bf16 (head-transposed); ctx: [b][s][h*128+e] f32
__global__ void attn_flash(const bf16* __restrict__ Q, const bf16* __restrict__ Km,
                           const bf16* __restrict__ Vt, float* __restrict__ ctx) {
  const int wave = threadIdx.x >> 5, lane = threadIdx.x & 31;
  const int gw = blockIdx.x * 4 + wave;      // 0..8191
  const int qt = gw & 31;
  const int bh = gw >> 5;                    // b*8 + h
  const int laneR = lane & 15;
  const int hi = (lane >= 16);
  const float scale = 0.08838834764831845f;  // 1/sqrt(128)

  const bf16* Qb = Q  + (size_t)bh * Sn * HDn;
  const bf16* Kb = Km + (size_t)bh * Sn * HDn;
  const bf16* Vb = Vt + (size_t)bh * HDn * Sn;

  ABv q[4];
  const bf16* qrow = Qb + (size_t)(qt * 16 + laneR) * HDn + (hi ? 8 : 0);
#pragma unroll
  for (int e = 0; e < 4; e++) {
    q[e].q[0] = *(const u32x4*)(qrow + e * 32);
    q[e].q[1] = *(const u32x4*)(qrow + e * 32 + 16);
  }

  v8f oc[8];
  const v8f vz = {0.f,0.f,0.f,0.f,0.f,0.f,0.f,0.f};
#pragma unroll
  for (int i = 0; i < 8; i++) oc[i] = vz;

  __shared__ float sP[4][16][32];
  __shared__ float sM[4][16];
  __shared__ float sL[4][16];
  __shared__ float sF[4][16];
  if (lane < 16) { sM[wave][lane] = -3e38f; sL[wave][lane] = 0.f; }

  for (int kt = 0; kt < 16; kt++) {
    __syncthreads();
    // scores for 32 keys (two 16x16 WMMA tiles), K-dim = HD = 128
#pragma unroll
    for (int sub = 0; sub < 2; sub++) {
      v8f sacc = vz;
      const bf16* krow = Kb + (size_t)(kt * 32 + sub * 16 + laneR) * HDn + (hi ? 8 : 0);
#pragma unroll
      for (int e = 0; e < 4; e++) {
        ABv bk;
        bk.q[0] = *(const u32x4*)(krow + e * 32);
        bk.q[1] = *(const u32x4*)(krow + e * 32 + 16);
        sacc = __builtin_amdgcn_wmma_f32_16x16x32_bf16(false, q[e].v, false, bk.v, (short)0, sacc, false, false);
      }
#pragma unroll
      for (int i = 0; i < 8; i++)
        sP[wave][i + (hi ? 8 : 0)][sub * 16 + laneR] = sacc[i] * scale;
    }
    __syncthreads();
    // online softmax (one lane per row)
    if (lane < 16) {
      int r = lane;
      float mold = sM[wave][r];
      float mx = mold;
      for (int j = 0; j < 32; j++) mx = fmaxf(mx, sP[wave][r][j]);
      float fac = __expf(mold - mx);
      float sum = sL[wave][r] * fac;
      for (int j = 0; j < 32; j++) {
        float p = __expf(sP[wave][r][j] - mx);
        sP[wave][r][j] = p;
        sum += p;
      }
      sM[wave][r] = mx; sL[wave][r] = sum; sF[wave][r] = fac;
    }
    __syncthreads();
    // rescale accumulators
#pragma unroll
    for (int i = 0; i < 8; i++) {
      float f = sF[wave][i + (hi ? 8 : 0)];
#pragma unroll
      for (int nt = 0; nt < 8; nt++) oc[nt][i] *= f;
    }
    // build P as bf16 A-operand (16x32)
    ABv pa;
    int ks = hi ? 8 : 0;
#pragma unroll
    for (int j = 0; j < 8; j++) {
      pa.v[j]     = f2bf(sP[wave][laneR][ks + j]);
      pa.v[8 + j] = f2bf(sP[wave][laneR][ks + 16 + j]);
    }
    // P @ V : 8 n-tiles over HD=128, K = 32 keys
#pragma unroll
    for (int nt = 0; nt < 8; nt++) {
      const bf16* vrow = Vb + (size_t)(nt * 16 + laneR) * Sn + kt * 32 + (hi ? 8 : 0);
      ABv bv;
      bv.q[0] = *(const u32x4*)(vrow);
      bv.q[1] = *(const u32x4*)(vrow + 16);
      oc[nt] = __builtin_amdgcn_wmma_f32_16x16x32_bf16(false, pa.v, false, bv.v, (short)0, oc[nt], false, false);
    }
  }

  const int b = bh >> 3, h = bh & 7;
#pragma unroll
  for (int i = 0; i < 8; i++) {
    int r = i + (hi ? 8 : 0);
    float inv = 1.f / sL[wave][r];
    int s = qt * 16 + r;
#pragma unroll
    for (int nt = 0; nt < 8; nt++)
      ctx[((size_t)b * Sn + s) * DM + h * HDn + nt * 16 + laneR] = oc[nt][i] * inv;
  }
}

// ---------------- mean pool + FC ----------------
__global__ void pool_mean(const float* __restrict__ ctx, float* __restrict__ pooled) {
  int idx = blockIdx.x * blockDim.x + threadIdx.x;
  if (idx >= Bn * DM) return;
  int b = idx >> 10, cc = idx & 1023;
  const float* p = ctx + (size_t)b * Sn * DM + cc;
  float s = 0.f;
  for (int t = 0; t < Sn; t++) s += p[(size_t)t * DM];
  pooled[idx] = s * (1.f / Sn);
}

__global__ void fc_out(const float* __restrict__ pooled, const float* __restrict__ Wfc,
                       const float* __restrict__ bfc, float* __restrict__ out) {
  int tid = threadIdx.x;
  if (tid >= Bn * 3) return;
  int b = tid / 3, o = tid % 3;
  float s = bfc[o];
  const float* pp = pooled + (size_t)b * DM;
  const float* w  = Wfc + (size_t)o * DM;
  for (int cIdx = 0; cIdx < DM; cIdx++) s += pp[cIdx] * w[cIdx];
  out[tid] = s;
}

// ---------------- host ----------------
extern "C" void kernel_launch(void* const* d_in, const int* in_sizes, int n_in,
                              void* d_out, int out_size, void* d_ws, size_t ws_size,
                              hipStream_t stream) {
  (void)in_sizes; (void)n_in; (void)out_size; (void)ws_size;
  const int*   xin = (const int*)d_in[0];
  const float* emb = (const float*)d_in[1];
  const float* Wih_in[2][2] = {{(const float*)d_in[2],  (const float*)d_in[6]},
                               {(const float*)d_in[10], (const float*)d_in[14]}};
  const float* Whh_in[2][2] = {{(const float*)d_in[3],  (const float*)d_in[7]},
                               {(const float*)d_in[11], (const float*)d_in[15]}};
  const float* bih_in[2][2] = {{(const float*)d_in[4],  (const float*)d_in[8]},
                               {(const float*)d_in[12], (const float*)d_in[16]}};
  const float* bhh_in[2][2] = {{(const float*)d_in[5],  (const float*)d_in[9]},
                               {(const float*)d_in[13], (const float*)d_in[17]}};
  const float* WqIn = (const float*)d_in[18]; const float* bqIn = (const float*)d_in[19];
  const float* WkIn = (const float*)d_in[20]; const float* bkIn = (const float*)d_in[21];
  const float* WvIn = (const float*)d_in[22]; const float* bvIn = (const float*)d_in[23];
  const float* WfcIn = (const float*)d_in[24]; const float* bfcIn = (const float*)d_in[25];
  float* out = (float*)d_out;

  char* ws = (char*)d_ws;
  size_t off = 0;
  auto alloc = [&](size_t bytes) -> char* {
    char* p = ws + off;
    off += (bytes + 255) & ~(size_t)255;
    return p;
  };

  bf16*  X0        = (bf16*)alloc((size_t)BSn * K0P * 2);
  bf16*  WihL0[2]  = {(bf16*)alloc((size_t)H4 * K0P * 2), (bf16*)alloc((size_t)H4 * K0P * 2)};
  bf16*  WihL1[2]  = {(bf16*)alloc((size_t)H4 * DM * 2),  (bf16*)alloc((size_t)H4 * DM * 2)};
  bf16*  WhhP[2][2]; float* biasP[2][2];
  for (int l = 0; l < 2; l++)
    for (int d = 0; d < 2; d++) {
      WhhP[l][d]  = (bf16*)alloc((size_t)H4 * Hn * 2);
      biasP[l][d] = (float*)alloc((size_t)H4 * 4);
    }
  float* xg[2] = {(float*)alloc((size_t)Sn * Bn * H4 * 4), (float*)alloc((size_t)Sn * Bn * H4 * 4)};
  bf16*  L0out = (bf16*)alloc((size_t)BSn * DM * 2);
  bf16*  L1out = (bf16*)alloc((size_t)BSn * DM * 2);
  bf16*  hbuf  = (bf16*)alloc((size_t)2 * 2 * Bn * Hn * 2);   // 131072 B
  float* cbuf  = (float*)alloc((size_t)2 * Bn * Hn * 4);      // 131072 B, contiguous after hbuf
  bf16*  Wattn[3] = {(bf16*)alloc((size_t)DM * DM * 2), (bf16*)alloc((size_t)DM * DM * 2),
                     (bf16*)alloc((size_t)DM * DM * 2)};
  bf16*  Qm  = (bf16*)alloc((size_t)Bn * NHn * Sn * HDn * 2);
  bf16*  Kmm = (bf16*)alloc((size_t)Bn * NHn * Sn * HDn * 2);
  bf16*  Vtt = (bf16*)alloc((size_t)Bn * NHn * Sn * HDn * 2);
  float* ctx = (float*)alloc((size_t)Bn * Sn * DM * 4);
  float* pooled = (float*)alloc((size_t)Bn * DM * 4);

  // 1) weight prep
  for (int l = 0; l < 2; l++)
    for (int d = 0; d < 2; d++) {
      int Kin  = l ? DM : Dn;
      int Kpad = l ? DM : K0P;
      bf16* wihO = l ? WihL1[d] : WihL0[d];
      prep_lstm<<<2048, 256, 0, stream>>>(Wih_in[l][d], Whh_in[l][d], bih_in[l][d], bhh_in[l][d],
                                          wihO, WhhP[l][d], biasP[l][d], Kin, Kpad);
    }
  prep_attn<<<1024, 256, 0, stream>>>(WqIn, Wattn[0]);
  prep_attn<<<1024, 256, 0, stream>>>(WkIn, Wattn[1]);
  prep_attn<<<1024, 256, 0, stream>>>(WvIn, Wattn[2]);

  // 2) embedding
  embed_gather<<<4096, 256, 0, stream>>>(xin, emb, X0);

  // 3) layer 0: input projections, then the sequential scan
  {
    dim3 g((BSn / 32) * (H4 / 64));
    gemm_wmma<<<g, 256, 0, stream>>>(X0, K0P, WihL0[0], K0P, biasP[0][0], xg[0], nullptr, H4, K0P, 1);
    gemm_wmma<<<g, 256, 0, stream>>>(X0, K0P, WihL0[1], K0P, biasP[0][1], xg[1], nullptr, H4, K0P, 1);
  }
  zero_f32<<<256, 256, 0, stream>>>((float*)hbuf, 65536);   // hbuf + cbuf (contiguous)
  for (int t = 0; t < Sn; t++)
    lstm_step<<<64, 128, 0, stream>>>(t, xg[0], xg[1], WhhP[0][0], WhhP[0][1], hbuf, cbuf, L0out);

  // 4) layer 1
  {
    dim3 g((BSn / 32) * (H4 / 64));
    gemm_wmma<<<g, 256, 0, stream>>>(L0out, DM, WihL1[0], DM, biasP[1][0], xg[0], nullptr, H4, DM, 1);
    gemm_wmma<<<g, 256, 0, stream>>>(L0out, DM, WihL1[1], DM, biasP[1][1], xg[1], nullptr, H4, DM, 1);
  }
  zero_f32<<<256, 256, 0, stream>>>((float*)hbuf, 65536);
  for (int t = 0; t < Sn; t++)
    lstm_step<<<64, 128, 0, stream>>>(t, xg[0], xg[1], WhhP[1][0], WhhP[1][1], hbuf, cbuf, L1out);

  // 5) QKV projections (Q,K -> [bh][s][e]; V -> [bh][e][s])
  {
    dim3 g((BSn / 32) * (DM / 64));
    gemm_wmma<<<g, 256, 0, stream>>>(L1out, DM, Wattn[0], DM, bqIn, nullptr, Qm,  DM, DM, 2);
    gemm_wmma<<<g, 256, 0, stream>>>(L1out, DM, Wattn[1], DM, bkIn, nullptr, Kmm, DM, DM, 2);
    gemm_wmma<<<g, 256, 0, stream>>>(L1out, DM, Wattn[2], DM, bvIn, nullptr, Vtt, DM, DM, 3);
  }

  // 6) flash attention
  attn_flash<<<2048, 128, 0, stream>>>(Qm, Kmm, Vtt, ctx);

  // 7) pool + fc
  pool_mean<<<(Bn * DM) / 256, 256, 0, stream>>>(ctx, pooled);
  fc_out<<<1, 96, 0, stream>>>(pooled, WfcIn, bfcIn, out);
}
